// MyRNNCell_27049704030527
// MI455X (gfx1250) — matrix operand order repaired and checked by reading, not compiled, not run on hardware
//
#include <hip/hip_runtime.h>
#include <math.h>

typedef __bf16 bf16;
typedef __attribute__((ext_vector_type(16))) __bf16 v16bf;
typedef __attribute__((ext_vector_type(8)))  __bf16 bf16x8;
typedef __attribute__((ext_vector_type(8)))  float  v8f;
typedef __attribute__((ext_vector_type(4)))  float  f32x4;

#define BATCH   16384
#define NOUT    1024
#define KIN     1024     // x features; hidden adds another 1024
#define KTOT    2048
#define TILE_M  128      // block tile M
#define TILE_N  256      // block tile N (8 waves as 2x4, each owning 64x64)
#define TILE_K  32
#define KSTEPS  (KTOT / TILE_K)
#define LDS_STR 40       // 32 bf16 + 16B pad per row -> bank-conflict-free

union FragU { v16bf v; struct { bf16x8 lo, hi; } p; };

static __device__ inline bf16x8 pack8(f32x4 a, f32x4 b) {
    bf16x8 r;
#pragma unroll
    for (int i = 0; i < 4; ++i) { r[i] = (bf16)a[i]; r[i + 4] = (bf16)b[i]; }
    return r;
}

__global__ __launch_bounds__(256, 1)
void rnncell_bf16_wmma(const float* __restrict__ x,
                       const float* __restrict__ hid,
                       const float* __restrict__ W,
                       const float* __restrict__ bias,
                       float* __restrict__ out)
{
    __shared__ bf16 ldsA[2][TILE_M * LDS_STR];   // 2 x 10 KB
    __shared__ bf16 ldsB[2][TILE_N * LDS_STR];   // 2 x 20 KB

    const int tid  = threadIdx.x;
    const int lane = tid & 31;
    const int wave = tid >> 5;

    const int blockN = blockIdx.x;   // 0..3   (out-feature tiles of 256)
    const int blockM = blockIdx.y;   // 0..127 (batch tiles of 128)

    // ---- producer mapping ------------------------------------------------
    // A: 128 rows x 32 k  -> each thread: 1 row-half (16 f32)
    // B: 256 rows x 32 k  -> each thread: 2 row-halves (rows r and r+128)
    const int ldRow  = tid >> 1;          // 0..127
    const int ldHalf = (tid & 1) * 16;    // 0 or 16 within the 32-wide K slice
    const size_t gmRow  = (size_t)blockM * TILE_M + ldRow;  // batch row (A)
    const size_t gnRow0 = (size_t)blockN * TILE_N + ldRow;  // W row (B, first)

    // ---- consumer mapping: wave -> 64x64 output sub-tile (2 waves M, 4 N)
    const int waveM = (wave >> 2) * 64;
    const int waveN = (wave & 3) * 64;

    v8f acc[4][4];
#pragma unroll
    for (int i = 0; i < 4; ++i)
#pragma unroll
        for (int j = 0; j < 4; ++j) acc[i][j] = v8f{};

    f32x4 aReg[4], bReg[8];

    auto loadG = [&](int it) {
        const int kbase = it * TILE_K;
        // concat(x, hidden) via pointer select: a K-step never straddles 1024
        const float* aSrc = (kbase < KIN)
            ? (x   + gmRow * (size_t)KIN + kbase          + ldHalf)
            : (hid + gmRow * (size_t)KIN + (kbase - KIN)  + ldHalf);
        const float* bSrc0 = W + gnRow0 * (size_t)KTOT + kbase + ldHalf;
        const float* bSrc1 = bSrc0 + (size_t)128 * KTOT;
#pragma unroll
        for (int i = 0; i < 4; ++i) aReg[i]     = ((const f32x4*)aSrc)[i];
#pragma unroll
        for (int i = 0; i < 4; ++i) bReg[i]     = ((const f32x4*)bSrc0)[i];
#pragma unroll
        for (int i = 0; i < 4; ++i) bReg[4 + i] = ((const f32x4*)bSrc1)[i];
    };

    auto storeLDS = [&](int buf) {
        bf16* arow  = &ldsA[buf][ldRow * LDS_STR + ldHalf];
        bf16* brow0 = &ldsB[buf][ldRow * LDS_STR + ldHalf];
        bf16* brow1 = brow0 + 128 * LDS_STR;
        *(bf16x8*)(arow)      = pack8(aReg[0], aReg[1]);   // ds_store_b128 x6
        *(bf16x8*)(arow  + 8) = pack8(aReg[2], aReg[3]);
        *(bf16x8*)(brow0)     = pack8(bReg[0], bReg[1]);
        *(bf16x8*)(brow0 + 8) = pack8(bReg[2], bReg[3]);
        *(bf16x8*)(brow1)     = pack8(bReg[4], bReg[5]);
        *(bf16x8*)(brow1 + 8) = pack8(bReg[6], bReg[7]);
    };

    // A fragment, ISA 16-bit 16x32 layout: lanes 0-15 -> K {0..7, 16..23},
    // lanes 16-31 -> K {8..15, 24..31}, row M = lane & 15
    auto loadFragA = [&](int buf, int mt) -> v16bf {
        const int row = waveM + mt * 16 + (lane & 15);
        const bf16* p = &ldsA[buf][row * LDS_STR + ((lane & 16) ? 8 : 0)];
        FragU f;
        f.p.lo = *(const bf16x8*)p;          // ds_load_b128
        f.p.hi = *(const bf16x8*)(p + 16);   // ds_load_b128
        return f.v;
    };

    // B fragment, 32x16 (KxN): lanes 0-15 -> K {0..7, 16..23},
    // lanes 16-31 -> K {8..15, 24..31}, N = lane & 15
    auto loadFragB = [&](int buf, int nt) -> v16bf {
        const int row = waveN + nt * 16 + (lane & 15);
        const bf16* p = &ldsB[buf][row * LDS_STR + ((lane & 16) ? 8 : 0)];
        FragU f;
        f.p.lo = *(const bf16x8*)p;
        f.p.hi = *(const bf16x8*)(p + 16);
        return f.v;
    };

    // ---- software-pipelined main loop, double-buffered LDS, 1 barrier/iter
    int buf = 0;
    loadG(0);
#pragma unroll 2
    for (int it = 0; it < KSTEPS; ++it) {
        storeLDS(buf);
        __syncthreads();
        if (it + 1 < KSTEPS) loadG(it + 1);   // overlap next global load with math

        v16bf aF[4], bF[4];
#pragma unroll
        for (int mt = 0; mt < 4; ++mt) aF[mt] = loadFragA(buf, mt);
#pragma unroll
        for (int nt = 0; nt < 4; ++nt) bF[nt] = loadFragB(buf, nt);

        // 16 WMMAs per 16 ds_load_b128: 1.5x more matrix work per LDS byte
#pragma unroll
        for (int mt = 0; mt < 4; ++mt)
#pragma unroll
            for (int nt = 0; nt < 4; ++nt)
                acc[mt][nt] = __builtin_amdgcn_wmma_f32_16x16x32_bf16(
                    /*neg_a=*/false, aF[mt], /*neg_b=*/false, bF[nt],
                    /*c_mod=*/(short)0, acc[mt][nt],
                    /*reuse_a=*/false, /*reuse_b=*/false);
        buf ^= 1;
    }

    // ---- epilogue: bias + tanh; C/D layout: M = vgpr + 8*(lane>=16), N = lane&15
    const size_t outRowBase = (size_t)blockM * TILE_M + waveM + ((lane >> 4) << 3);
    const int ncol = lane & 15;
#pragma unroll
    for (int nt = 0; nt < 4; ++nt) {
        const int col = blockN * TILE_N + waveN + nt * 16 + ncol;
        const float bv = bias[col];
#pragma unroll
        for (int mt = 0; mt < 4; ++mt) {
#pragma unroll
            for (int v = 0; v < 8; ++v) {
                const size_t row = outRowBase + mt * 16 + v;
                out[row * NOUT + col] = tanhf(acc[mt][nt][v] + bv);
            }
        }
    }
}

extern "C" void kernel_launch(void* const* d_in, const int* in_sizes, int n_in,
                              void* d_out, int out_size, void* d_ws, size_t ws_size,
                              hipStream_t stream) {
    (void)in_sizes; (void)n_in; (void)out_size; (void)d_ws; (void)ws_size;
    const float* x   = (const float*)d_in[0];
    const float* hid = (const float*)d_in[1];
    const float* W   = (const float*)d_in[2];
    const float* b   = (const float*)d_in[3];
    float* out = (float*)d_out;

    dim3 grid(NOUT / TILE_N, BATCH / TILE_M);   // (4, 128)
    rnncell_bf16_wmma<<<grid, 256, 0, stream>>>(x, hid, W, b, out);
}
